// GroupedQueryAttention_33809982554659
// MI455X (gfx1250) — matrix operand order repaired and checked by reading, not compiled, not run on hardware
//
#include <hip/hip_runtime.h>
#include <hip/hip_bf16.h>

// ---------------- problem constants ----------------
#define BATCH 2
#define SEQ   2048
#define HID   2048
#define NH    16
#define NKV   4
#define HD    128
#define GROUPS (NH / NKV)
#define MROWS (BATCH * SEQ)   // 4096

typedef __attribute__((ext_vector_type(16))) __bf16 bf16x16;
typedef __attribute__((ext_vector_type(8)))  float  f32x8;

union Frag {
    bf16x16 v;
    uint4   q[2];
};

// CDNA5 async memory->LDS copy (GV mode: per-lane LDS offset + 64-bit gaddr),
// tracked by ASYNCcnt.
__device__ __forceinline__ void async_copy_b128(unsigned lds_off,
                                                const void* gaddr) {
    unsigned long long ga = (unsigned long long)(uintptr_t)gaddr;
    asm volatile("global_load_async_to_lds_b128 %0, %1, off"
                 :: "v"(lds_off), "v"(ga) : "memory");
}
__device__ __forceinline__ void wait_asynccnt0() {
    asm volatile("s_wait_asynccnt 0" ::: "memory");
}

// ---------------------------------------------------------------------------
// 1) fp32 -> bf16 elementwise convert
// ---------------------------------------------------------------------------
__global__ void cvt_bf16_kernel(const float* __restrict__ in,
                                __bf16* __restrict__ out, int n) {
    int i = blockIdx.x * 256 + threadIdx.x;
    if (i < n) out[i] = (__bf16)in[i];
}

// ---------------------------------------------------------------------------
// 2) fp32 W[K,N] -> bf16 Wt[N,K]   (so GEMM B-operand is K-contiguous)
// ---------------------------------------------------------------------------
__global__ void cvt_transpose_kernel(const float* __restrict__ W,
                                     __bf16* __restrict__ Wt, int K, int N) {
    int i = blockIdx.x * 256 + threadIdx.x;
    if (i >= K * N) return;
    int k = i / N, n = i % N;
    Wt[(size_t)n * K + k] = (__bf16)W[i];
}

// ---------------------------------------------------------------------------
// 3) Tiled WMMA GEMM: C[M,N] (f32) = A[M,K] (bf16, row-major) * Bt[N,K]^T
//    Block = 256 threads = 8 waves. Tile 128x128, K-step 32.
//    Double-buffered async-DMA staging: DMA(i+1) overlaps WMMA(i).
//    Row pad 32->40 elems (80B = 20-word stride) kills LDS bank conflicts.
// ---------------------------------------------------------------------------
__global__ __launch_bounds__(256)
void gemm_bf16_wmma(const __bf16* __restrict__ A,
                    const __bf16* __restrict__ Bt,
                    float* __restrict__ C,
                    int Mdim, int Ndim, int Kdim) {
    __shared__ __bf16 As[2][128][40];   // 2 x 10 KB
    __shared__ __bf16 Bs[2][128][40];   // 2 x 10 KB (rows = output columns)

    const int tid  = threadIdx.x;
    const int lane = tid & 31;
    const int wave = tid >> 5;
    const int lo   = lane & 15;
    const int hi   = lane >> 4;
    const int m0   = blockIdx.x * 128;
    const int n0   = blockIdx.y * 128;

    // this thread's two fixed staging chunks (16B each, 512 chunks per tile)
    const int rowA0 = tid >> 1,          c40 = (tid & 1) * 2;      // chunks 0,1
    const int rowA1 = tid >> 1,          c41 = (tid & 1) * 2 + 1;  // interleave
    const __bf16* pa0 = A  + (size_t)(m0 + rowA0) * Kdim + c40 * 8;
    const __bf16* pa1 = A  + (size_t)(m0 + rowA1) * Kdim + c41 * 8;
    const __bf16* pb0 = Bt + (size_t)(n0 + rowA0) * Kdim + c40 * 8;
    const __bf16* pb1 = Bt + (size_t)(n0 + rowA1) * Kdim + c41 * 8;

    f32x8 acc[8];
#pragma unroll
    for (int i = 0; i < 8; ++i)
        acc[i] = (f32x8){0.f, 0.f, 0.f, 0.f, 0.f, 0.f, 0.f, 0.f};

    auto stage = [&](int k0, int buf) {
        async_copy_b128((unsigned)(uintptr_t)&As[buf][rowA0][c40 * 8], pa0 + k0);
        async_copy_b128((unsigned)(uintptr_t)&As[buf][rowA1][c41 * 8], pa1 + k0);
        async_copy_b128((unsigned)(uintptr_t)&Bs[buf][rowA0][c40 * 8], pb0 + k0);
        async_copy_b128((unsigned)(uintptr_t)&Bs[buf][rowA1][c41 * 8], pb1 + k0);
    };

    stage(0, 0);
    int buf = 0;
    for (int k0 = 0; k0 < Kdim; k0 += 32, buf ^= 1) {
        wait_asynccnt0();                 // this tile's DMA done (ours)
        __syncthreads();                  // everyone's DMA done; prev reads done
        if (k0 + 32 < Kdim) stage(k0 + 32, buf ^ 1);   // overlap with compute

        // A fragment: lane -> row m = lane&15 of this wave's 16-row strip;
        // K(e) = 8*hi + (e&7) + 16*(e>>3)  => two contiguous 16B LDS loads
        Frag a;
        int arow = wave * 16 + lo;
        a.q[0] = *(const uint4*)(&As[buf][arow][8 * hi]);
        a.q[1] = *(const uint4*)(&As[buf][arow][16 + 8 * hi]);

        Frag bfr[8];
#pragma unroll
        for (int nb = 0; nb < 8; ++nb) {
            int brow = nb * 16 + lo;
            bfr[nb].q[0] = *(const uint4*)(&Bs[buf][brow][8 * hi]);
            bfr[nb].q[1] = *(const uint4*)(&Bs[buf][brow][16 + 8 * hi]);
        }
#pragma unroll
        for (int nb = 0; nb < 8; ++nb)
            acc[nb] = __builtin_amdgcn_wmma_f32_16x16x32_bf16(
                false, a.v, false, bfr[nb].v, (short)0, acc[nb], false, false);
    }

    // store: lane holds col nb*16+lo; VGPR r holds row wave*16 + r + 8*hi
#pragma unroll
    for (int nb = 0; nb < 8; ++nb) {
        int col = n0 + nb * 16 + lo;
#pragma unroll
        for (int r = 0; r < 8; ++r) {
            int row = m0 + wave * 16 + r + 8 * hi;
            C[(size_t)row * Ndim + col] = acc[nb][r];
        }
    }
}

// ---------------------------------------------------------------------------
// 4) RoPE + repack:
//    qf[M,2048] -> qb [B,NH,T,HD]  bf16 (rotary applied)
//    kf[M, 512] -> kb [B,NKV,T,HD] bf16 (rotary applied)
//    vf[M, 512] -> vt [B,NKV,HD,T] bf16 (transposed for PV WMMA B-fragments)
// ---------------------------------------------------------------------------
__global__ void rope_pack_kernel(const float* __restrict__ qf,
                                 const float* __restrict__ kf,
                                 const float* __restrict__ vf,
                                 const float* __restrict__ cosT,
                                 const float* __restrict__ sinT,
                                 __bf16* __restrict__ qb,
                                 __bf16* __restrict__ kb,
                                 __bf16* __restrict__ vt) {
    int idx = blockIdx.x * 256 + threadIdx.x;          // over M * 3072
    int col = idx % 3072;
    int row = idx / 3072;                               // row = b*T + t
    int t   = row % SEQ;
    int b   = row / SEQ;

    if (col < 2048) {                                   // --- Q ---
        int h = col >> 7, d = col & 127;
        float v  = qf[(size_t)row * 2048 + col];
        int   dp = (d < 64) ? d + 64 : d - 64;
        float pr = qf[(size_t)row * 2048 + (h << 7) + dp];
        float rt = (d < 64) ? -pr : pr;
        float o  = v * cosT[t * HD + d] + rt * sinT[t * HD + d];
        qb[(((size_t)(b * NH + h)) * SEQ + t) * HD + d] = (__bf16)o;
    } else if (col < 2560) {                            // --- K ---
        int c = col - 2048;
        int h = c >> 7, d = c & 127;
        float v  = kf[(size_t)row * 512 + c];
        int   dp = (d < 64) ? d + 64 : d - 64;
        float pr = kf[(size_t)row * 512 + (h << 7) + dp];
        float rt = (d < 64) ? -pr : pr;
        float o  = v * cosT[t * HD + d] + rt * sinT[t * HD + d];
        kb[(((size_t)(b * NKV + h)) * SEQ + t) * HD + d] = (__bf16)o;
    } else {                                            // --- V (transpose) ---
        int c = col - 2560;
        int h = c >> 7, d = c & 127;
        vt[(((size_t)(b * NKV + h)) * HD + d) * SEQ + t] =
            (__bf16)vf[(size_t)row * 512 + c];
    }
}

// ---------------------------------------------------------------------------
// 5) Causal flash attention, WMMA QK^T and PV.
//    Grid: (B*NH, T/64). Block: 128 threads = 4 waves; wave owns 16 q-rows.
//    K/V chunks staged into LDS by double-buffered async DMA (shared by the
//    4 waves -> 4x less global traffic than per-wave loads).
// ---------------------------------------------------------------------------
__global__ __launch_bounds__(128)
void attn_wmma_kernel(const __bf16* __restrict__ Qb,   // [B,NH,T,HD]
                      const __bf16* __restrict__ Kb,   // [B,NKV,T,HD]
                      const __bf16* __restrict__ Vt,   // [B,NKV,HD,T]
                      __bf16* __restrict__ Ob) {       // [B*T, NH*HD]
    __shared__ __bf16 Ks[2][64][136];   // keys: row=s, 128 d (+8 pad)  ~34 KB
    __shared__ __bf16 Vs[2][128][72];   // V^T:  row=d,  64 s (+8 pad)  ~36 KB
    __shared__ __bf16 Pl[4][16][72];    // per-wave P tile (+8 pad)     ~9 KB

    const int tid  = threadIdx.x;
    const int lane = tid & 31;
    const int wave = tid >> 5;
    const int lo   = lane & 15;
    const int hi   = lane >> 4;

    const int h  = blockIdx.x % NH;
    const int b  = blockIdx.x / NH;
    const int hk = h / GROUPS;
    const int q0 = blockIdx.y * 64;
    const int qr = q0 + wave * 16;                     // this wave's first q row
    const float scale = 0.08838834764831845f;          // 1/sqrt(128)

    const __bf16* kbase = Kb + (size_t)(b * NKV + hk) * SEQ * HD;
    const __bf16* vbase = Vt + (size_t)(b * NKV + hk) * HD * SEQ;

    // ---- Q fragments (kept in registers for the whole key loop) ----
    const __bf16* qbase = Qb + ((size_t)(b * NH + h) * SEQ + qr) * HD;
    Frag aq[4];
#pragma unroll
    for (int kk = 0; kk < 4; ++kk) {
        aq[kk].q[0] = *(const uint4*)(qbase + (size_t)lo * HD + kk * 32 + 8 * hi);
        aq[kk].q[1] = *(const uint4*)(qbase + (size_t)lo * HD + kk * 32 + 16 + 8 * hi);
    }

    f32x8 o[8];
#pragma unroll
    for (int i = 0; i < 8; ++i)
        o[i] = (f32x8){0.f, 0.f, 0.f, 0.f, 0.f, 0.f, 0.f, 0.f};
    float run_m[8], run_l[8];
#pragma unroll
    for (int r = 0; r < 8; ++r) { run_m[r] = -3.0e38f; run_l[r] = 0.f; }

    // cooperative async stage of one 64-key K chunk + V chunk into LDS:
    // K: 64 rows x 256B  = 1024 x 16B chunks; V: 128 rows x 128B = 1024 chunks
    auto stage_kv = [&](int s0, int buf) {
#pragma unroll
        for (int j = 0; j < 8; ++j) {
            int ck  = tid + j * 128;            // 0..1023
            int krow = ck >> 4, kc = ck & 15;   // K: 16 chunks per row
            async_copy_b128((unsigned)(uintptr_t)&Ks[buf][krow][kc * 8],
                            kbase + (size_t)(s0 + krow) * HD + kc * 8);
            int vrow = ck >> 3, vc = ck & 7;    // V: 8 chunks per row
            async_copy_b128((unsigned)(uintptr_t)&Vs[buf][vrow][vc * 8],
                            vbase + (size_t)vrow * SEQ + s0 + vc * 8);
        }
    };

    stage_kv(0, 0);
    int buf = 0;
    for (int s0 = 0; s0 < q0 + 64; s0 += 64, buf ^= 1) {   // causal 64-key chunks
        wait_asynccnt0();
        __syncthreads();
        if (s0 + 64 < q0 + 64) stage_kv(s0 + 64, buf ^ 1); // overlap with compute

        // ---- scores S = Q K^T (16 x 64 per wave) ----
        f32x8 sc[4];
#pragma unroll
        for (int nb = 0; nb < 4; ++nb) {
            sc[nb] = (f32x8){0.f, 0.f, 0.f, 0.f, 0.f, 0.f, 0.f, 0.f};
            Frag bk[4];
#pragma unroll
            for (int kk = 0; kk < 4; ++kk) {
                bk[kk].q[0] = *(const uint4*)(&Ks[buf][nb * 16 + lo][kk * 32 + 8 * hi]);
                bk[kk].q[1] = *(const uint4*)(&Ks[buf][nb * 16 + lo][kk * 32 + 16 + 8 * hi]);
            }
#pragma unroll
            for (int kk = 0; kk < 4; ++kk)
                sc[nb] = __builtin_amdgcn_wmma_f32_16x16x32_bf16(
                    false, aq[kk].v, false, bk[kk].v, (short)0, sc[nb], false, false);
        }

        // ---- scale + causal mask ----
#pragma unroll
        for (int nb = 0; nb < 4; ++nb) {
            int sg = s0 + nb * 16 + lo;
#pragma unroll
            for (int r = 0; r < 8; ++r) {
                int mg = qr + r + 8 * hi;
                float v = sc[nb][r] * scale;
                sc[nb][r] = (sg <= mg) ? v : -1.0e30f;
            }
        }

        // ---- row max over this chunk (reduce across the 16-lane half) ----
        float mx[8];
#pragma unroll
        for (int r = 0; r < 8; ++r) {
            float m = sc[0][r];
            m = fmaxf(m, sc[1][r]);
            m = fmaxf(m, sc[2][r]);
            m = fmaxf(m, sc[3][r]);
            m = fmaxf(m, __shfl_xor(m, 8, 32));
            m = fmaxf(m, __shfl_xor(m, 4, 32));
            m = fmaxf(m, __shfl_xor(m, 2, 32));
            m = fmaxf(m, __shfl_xor(m, 1, 32));
            mx[r] = m;
        }

        // ---- online softmax update ----
        float alpha[8];
#pragma unroll
        for (int r = 0; r < 8; ++r) {
            float nm = fmaxf(run_m[r], mx[r]);
            alpha[r] = __expf(run_m[r] - nm);
            run_m[r] = nm;
        }
        float rs[8];
#pragma unroll
        for (int r = 0; r < 8; ++r) rs[r] = 0.f;
#pragma unroll
        for (int nb = 0; nb < 4; ++nb)
#pragma unroll
            for (int r = 0; r < 8; ++r) {
                float p = __expf(sc[nb][r] - run_m[r]);
                sc[nb][r] = p;
                rs[r] += p;
            }
#pragma unroll
        for (int r = 0; r < 8; ++r) {
            float s = rs[r];
            s += __shfl_xor(s, 8, 32);
            s += __shfl_xor(s, 4, 32);
            s += __shfl_xor(s, 2, 32);
            s += __shfl_xor(s, 1, 32);
            run_l[r] = run_l[r] * alpha[r] + s;
        }
#pragma unroll
        for (int db = 0; db < 8; ++db)
#pragma unroll
            for (int r = 0; r < 8; ++r) o[db][r] *= alpha[r];

        // ---- P (f32 D-layout) -> bf16 A-layout via LDS transpose ----
#pragma unroll
        for (int nb = 0; nb < 4; ++nb)
#pragma unroll
            for (int r = 0; r < 8; ++r)
                Pl[wave][r + 8 * hi][nb * 16 + lo] = (__bf16)sc[nb][r];
        __syncthreads();

        Frag ap[2];
#pragma unroll
        for (int k2 = 0; k2 < 2; ++k2) {
            ap[k2].q[0] = *(const uint4*)(&Pl[wave][lo][k2 * 32 + 8 * hi]);
            ap[k2].q[1] = *(const uint4*)(&Pl[wave][lo][k2 * 32 + 16 + 8 * hi]);
        }
        __syncthreads();

        // ---- O += P V  (V^T staged in LDS: contiguous s per lane) ----
#pragma unroll
        for (int db = 0; db < 8; ++db) {
            Frag bv[2];
#pragma unroll
            for (int k2 = 0; k2 < 2; ++k2) {
                bv[k2].q[0] = *(const uint4*)(&Vs[buf][db * 16 + lo][k2 * 32 + 8 * hi]);
                bv[k2].q[1] = *(const uint4*)(&Vs[buf][db * 16 + lo][k2 * 32 + 16 + 8 * hi]);
            }
#pragma unroll
            for (int k2 = 0; k2 < 2; ++k2)
                o[db] = __builtin_amdgcn_wmma_f32_16x16x32_bf16(
                    false, ap[k2].v, false, bv[k2].v, (short)0, o[db], false, false);
        }
    }

    // ---- normalize + store [B*T, NH*HD] bf16 ----
#pragma unroll
    for (int db = 0; db < 8; ++db) {
#pragma unroll
        for (int r = 0; r < 8; ++r) {
            int row = b * SEQ + qr + r + 8 * hi;
            Ob[(size_t)row * (NH * HD) + h * HD + db * 16 + lo] =
                (__bf16)(o[db][r] / run_l[r]);
        }
    }
}

// ---------------------------------------------------------------------------
// Host-side orchestration
// ---------------------------------------------------------------------------
extern "C" void kernel_launch(void* const* d_in, const int* in_sizes, int n_in,
                              void* d_out, int out_size, void* d_ws, size_t ws_size,
                              hipStream_t stream) {
    const float* x    = (const float*)d_in[0];
    const float* cosT = (const float*)d_in[1];   // [1,1,T,HD]
    const float* sinT = (const float*)d_in[2];
    const float* Wq   = (const float*)d_in[3];   // [HID, NH*HD]
    const float* Wk   = (const float*)d_in[4];   // [HID, NKV*HD]
    const float* Wv   = (const float*)d_in[5];
    const float* Wo   = (const float*)d_in[6];   // [NH*HD, HID]
    float* out = (float*)d_out;

    char*  ws  = (char*)d_ws;
    size_t off = 0;
    auto alloc = [&](size_t bytes) -> void* {
        void* p = ws + off;
        off += (bytes + 255) & ~(size_t)255;
        return p;
    };

    __bf16* xb   = (__bf16*)alloc((size_t)MROWS * HID * 2);     // x in bf16
    __bf16* Wqt  = (__bf16*)alloc((size_t)2048 * 2048 * 2);     // Wq^T bf16
    __bf16* Wkt  = (__bf16*)alloc((size_t)512  * 2048 * 2);
    __bf16* Wvt  = (__bf16*)alloc((size_t)512  * 2048 * 2);
    __bf16* Wot  = (__bf16*)alloc((size_t)2048 * 2048 * 2);     // Wo^T bf16
    float*  qf   = (float*) alloc((size_t)MROWS * 2048 * 4);
    float*  kf   = (float*) alloc((size_t)MROWS * 512  * 4);
    float*  vf   = (float*) alloc((size_t)MROWS * 512  * 4);
    __bf16* qb   = (__bf16*)alloc((size_t)MROWS * 2048 * 2);    // [B,NH,T,HD]
    __bf16* kb   = (__bf16*)alloc((size_t)MROWS * 512  * 2);    // [B,NKV,T,HD]
    __bf16* vt   = (__bf16*)alloc((size_t)MROWS * 512  * 2);    // [B,NKV,HD,T]
    __bf16* ab   = (__bf16*)alloc((size_t)MROWS * 2048 * 2);    // attn out bf16

    // 1) converts
    cvt_bf16_kernel<<<(MROWS * HID) / 256, 256, 0, stream>>>(x, xb, MROWS * HID);
    cvt_transpose_kernel<<<(2048 * 2048) / 256, 256, 0, stream>>>(Wq, Wqt, 2048, 2048);
    cvt_transpose_kernel<<<(2048 * 512 ) / 256, 256, 0, stream>>>(Wk, Wkt, 2048, 512);
    cvt_transpose_kernel<<<(2048 * 512 ) / 256, 256, 0, stream>>>(Wv, Wvt, 2048, 512);
    cvt_transpose_kernel<<<(2048 * 2048) / 256, 256, 0, stream>>>(Wo, Wot, 2048, 2048);

    // 2) QKV projections (WMMA GEMMs)
    gemm_bf16_wmma<<<dim3(MROWS / 128, 2048 / 128), 256, 0, stream>>>(
        xb, Wqt, qf, MROWS, 2048, 2048);
    gemm_bf16_wmma<<<dim3(MROWS / 128, 512 / 128), 256, 0, stream>>>(
        xb, Wkt, kf, MROWS, 512, 2048);
    gemm_bf16_wmma<<<dim3(MROWS / 128, 512 / 128), 256, 0, stream>>>(
        xb, Wvt, vf, MROWS, 512, 2048);

    // 3) RoPE + repack to attention-friendly bf16 layouts
    rope_pack_kernel<<<(MROWS * 3072) / 256, 256, 0, stream>>>(
        qf, kf, vf, cosT, sinT, qb, kb, vt);

    // 4) causal GQA flash attention (WMMA, async K/V staging)
    attn_wmma_kernel<<<dim3(BATCH * NH, SEQ / 64), 128, 0, stream>>>(qb, kb, vt, ab);

    // 5) output projection (WMMA GEMM, f32 out)
    gemm_bf16_wmma<<<dim3(MROWS / 128, 2048 / 128), 256, 0, stream>>>(
        ab, Wot, out, MROWS, 2048, 2048);
}
